// EncoderLayer_6167573037301
// MI455X (gfx1250) — compile-verified, matrix operand order checked by reading, not run on hardware
//
#include <hip/hip_runtime.h>

typedef __attribute__((ext_vector_type(16))) __bf16 v16bf;
typedef __attribute__((ext_vector_type(8)))  float v8f;
typedef int v4i_async __attribute__((vector_size(16)));
using u16 = unsigned short;

#define BSZ 4
#define SEQ 512
#define DMODEL 1024
#define NH 16
#define DKH 64
#define FFD 4096
#define MROWS (BSZ*SEQ)   // 2048

#if __has_builtin(__builtin_amdgcn_global_load_async_to_lds_b128)
#define HAS_ASYNC 1
#else
#define HAS_ASYNC 0
#endif

__device__ __forceinline__ u16 f2bf(float f) {
    unsigned u = __float_as_uint(f);
    unsigned r = (u + 0x7FFFu + ((u >> 16) & 1u)) >> 16;
    return (u16)r;
}
__device__ __forceinline__ float bf2f(u16 h) {
    return __uint_as_float(((unsigned)h) << 16);
}

// 16-byte global->LDS copy: async on CDNA5 (tracked by ASYNCcnt), sync fallback.
__device__ __forceinline__ void cp16(void* l, const void* g) {
#if HAS_ASYNC
    __builtin_amdgcn_global_load_async_to_lds_b128(
        (__attribute__((address_space(1))) v4i_async*)g,
        (__attribute__((address_space(3))) v4i_async*)l,
        0, 0);
#else
    *(uint4*)l = *(const uint4*)g;
#endif
}
__device__ __forceinline__ void async_fence() {
#if HAS_ASYNC
#if __has_builtin(__builtin_amdgcn_s_wait_asynccnt)
    __builtin_amdgcn_s_wait_asynccnt(0);
#else
    asm volatile("s_wait_asynccnt 0x0" ::: "memory");
#endif
#endif
}

union Frag { v16bf v; unsigned u[8]; __device__ Frag() {} };

// ---------------------------------------------------------------- f32[K][N] -> bf16[N][K] (transpose-convert)
__global__ void f2bf_t_kernel(const float* __restrict__ src, u16* __restrict__ dst,
                              int K, int N) {
    __shared__ float tile[32][33];
    int k0 = blockIdx.x * 32, n0 = blockIdx.y * 32;
    int tx = threadIdx.x & 31, ty = threadIdx.x >> 5;   // 256 thr: ty 0..7
#pragma unroll
    for (int i = 0; i < 4; ++i)
        tile[ty + i * 8][tx] = src[(size_t)(k0 + ty + i * 8) * N + n0 + tx];
    __syncthreads();
#pragma unroll
    for (int i = 0; i < 4; ++i)
        dst[(size_t)(n0 + ty + i * 8) * K + k0 + tx] = f2bf(tile[tx][ty + i * 8]);
}

// ---------------------------------------------------------------- v2[b][s][c] -> vt[b][c][s]  (c = h*64+d)
__global__ void vtrans_kernel(const u16* __restrict__ v2, u16* __restrict__ vt) {
    __shared__ u16 t[32][33];
    int s0 = blockIdx.x * 32, c0 = blockIdx.y * 32, b = blockIdx.z;
    int tx = threadIdx.x & 31, ty = threadIdx.x >> 5;
#pragma unroll
    for (int i = 0; i < 4; ++i)
        t[ty + i * 8][tx] = v2[(size_t)(b * SEQ + s0 + ty + i * 8) * DMODEL + c0 + tx];
    __syncthreads();
#pragma unroll
    for (int i = 0; i < 4; ++i)
        vt[(size_t)(b * DMODEL + c0 + ty + i * 8) * SEQ + s0 + tx] = t[tx][ty + i * 8];
}

// ---------------------------------------------------------------- dual LayerNorm (bf16 out x2)
__global__ void ln_dual_kernel(const float* __restrict__ x,
                               const float* __restrict__ g0, const float* __restrict__ b0,
                               const float* __restrict__ g1, const float* __restrict__ b1,
                               u16* __restrict__ o0, u16* __restrict__ o1) {
    __shared__ float rs[256], rq[256];
    int row = blockIdx.x, tid = threadIdx.x;
    const float* xr = x + (size_t)row * DMODEL;
    float v[4], s = 0.f, q = 0.f;
#pragma unroll
    for (int i = 0; i < 4; ++i) { v[i] = xr[tid + i * 256]; s += v[i]; q += v[i] * v[i]; }
    rs[tid] = s; rq[tid] = q; __syncthreads();
    for (int st = 128; st > 0; st >>= 1) {
        if (tid < st) { rs[tid] += rs[tid + st]; rq[tid] += rq[tid + st]; }
        __syncthreads();
    }
    float mean = rs[0] * (1.f / DMODEL);
    float var  = rq[0] * (1.f / DMODEL) - mean * mean;
    float rstd = rsqrtf(var + 1e-5f);
#pragma unroll
    for (int i = 0; i < 4; ++i) {
        int c = tid + i * 256;
        float nm = (v[i] - mean) * rstd;
        o0[(size_t)row * DMODEL + c] = f2bf(nm * g0[c] + b0[c]);
        if (o1) o1[(size_t)row * DMODEL + c] = f2bf(nm * g1[c] + b1[c]);
    }
}

// ---------------------------------------------------------------- per-head q/k projection (64x64)
__global__ void qkproj_kernel(const u16* __restrict__ ctx,
                              const float* __restrict__ wq, const float* __restrict__ bq,
                              const float* __restrict__ wk, const float* __restrict__ bk,
                              float* __restrict__ qv, float* __restrict__ kv) {
    int t = blockIdx.x * 256 + threadIdx.x;      // [0, B*H*S*64)
    int d = t & 63, flat = t >> 6;               // flat = (b*16+h)*512 + s
    int s = flat & 511, h = (flat >> 9) & 15, b = flat >> 13;
    const u16* cr = ctx + (size_t)(b * SEQ + s) * DMODEL + h * DKH;
    float aq = bq[d], ak = bk[d];
#pragma unroll 8
    for (int k = 0; k < 64; ++k) {
        float cv = bf2f(cr[k]);
        aq += cv * wq[k * 64 + d];
        ak += cv * wk[k * 64 + d];
    }
    qv[(size_t)flat * 64 + d] = aq;
    kv[(size_t)flat * 64 + d] = ak;
}

// ---------------------------------------------------------------- band scores, softmax pair, prefix scan
__global__ void band_kernel(const float* __restrict__ qv, const float* __restrict__ kv,
                            const float* __restrict__ prior,
                            float* __restrict__ naB, float* __restrict__ Pg) {
    __shared__ float dn[512];
    __shared__ float eb[512];
    int bh = blockIdx.x, s = threadIdx.x;
    const float* qr = qv + ((size_t)bh * SEQ + s) * 64;
    float ds = 0.f, us = 0.f;
    if (s > 0) {
        const float* kr = kv + ((size_t)bh * SEQ + s - 1) * 64;
        float a = 0.f;
#pragma unroll 8
        for (int k = 0; k < 64; ++k) a += qr[k] * kr[k];
        ds = a * (1.f / 64.f);
    }
    if (s < 511) {
        const float* kr = kv + ((size_t)bh * SEQ + s + 1) * 64;
        float a = 0.f;
#pragma unroll 8
        for (int k = 0; k < 64; ++k) a += qr[k] * kr[k];
        us = a * (1.f / 64.f);
    }
    float nup, ndn;
    if (s == 0)        { nup = 1.f; ndn = 0.f; }
    else if (s == 511) { ndn = 1.f; nup = 0.f; }
    else {
        float m = fmaxf(ds, us);
        float ed = __expf(ds - m), eu = __expf(us - m), z = ed + eu;
        nup = eu / z; ndn = ed / z;
    }
    dn[s] = ndn; __syncthreads();
    float pr = prior[0];
    if (s < 511) {
        float nb = pr + (1.f - pr) * sqrtf(nup * dn[s + 1] + 1e-4f);
        naB[(size_t)bh * SEQ + s] = nb;
        eb[s + 1] = logf(nb + 1e-9f);
    }
    __syncthreads();
    if (s == 0) {
        float acc = 0.f;
        Pg[(size_t)bh * SEQ] = 0.f;
        for (int m = 1; m < 512; ++m) { acc += eb[m]; Pg[(size_t)bh * SEQ + m] = acc; }
    }
}

// ---------------------------------------------------------------- fill group_prob + na (memory bound)
__global__ void fill_kernel(const float* __restrict__ naB, const float* __restrict__ Pg,
                            const float* __restrict__ prior,
                            float* __restrict__ gpo, float* __restrict__ nao) {
    __shared__ float Ps[512];
    int bid = blockIdx.x;           // bh*512 + i
    int bh = bid >> 9, i = bid & 511, j = threadIdx.x;
    Ps[j] = Pg[(size_t)bh * SEQ + j];
    __syncthreads();
    float pr = prior[0];
    float offc = pr + (1.f - pr) * 0.01f;
    float gp;
    if (i == j) gp = offc;
    else {
        int lo = (i < j) ? i : j, hi2 = (i < j) ? j : i;
        int h1 = (hi2 + 1 < 511) ? hi2 + 1 : 511;
        float lg = Ps[h1] + Ps[hi2] - Ps[(lo > 0) ? lo - 1 : 0] - Ps[lo];
        gp = 1e-4f + __expf(lg);
    }
    float nv;
    if (j == i + 1)      nv = naB[(size_t)bh * SEQ + i];
    else if (j + 1 == i) nv = naB[(size_t)bh * SEQ + j];
    else                 nv = offc;
    size_t base = (size_t)bid * SEQ;
    gpo[base + j] = gp;
    nao[base + j] = nv;
}

// ---------------------------------------------------------------- bf16 WMMA GEMM, double-buffered async staging
// A[M,K] bf16 rm, Wt[N,K] bf16 (pre-transposed). Block 256 thr = 8 waves, tile 128x128, BK=32.
template <int RELU, int HAS_RES, int OUT_BF16>
__global__ void gemm_bf16(const u16* __restrict__ A, const u16* __restrict__ Wt,
                          const float* __restrict__ bias, const float* __restrict__ res,
                          void* __restrict__ out, int M, int N, int K) {
    __shared__ u16 As[2][128 * 32];   // [m][k]
    __shared__ u16 Bs[2][128 * 32];   // [n][k]
    int m0 = blockIdx.y * 128, n0 = blockIdx.x * 128;
    int tid = threadIdx.x, wave = tid >> 5, lane = tid & 31;
    int wm = wave >> 1, wn = wave & 1;
    int m16 = lane & 15, hi = lane >> 4;

    v8f acc[2][4];
#pragma unroll
    for (int r = 0; r < 2; ++r)
#pragma unroll
        for (int c = 0; c < 4; ++c) acc[r][c] = (v8f){0,0,0,0,0,0,0,0};

    auto stage = [&](int buf, int kk) {
#pragma unroll
        for (int it = 0; it < 2; ++it) {
            int seg = tid + it * 256;          // 512 segments of 8 bf16
            int row = seg >> 2, q8 = (seg & 3) * 8;
            cp16(&As[buf][row * 32 + q8], &A [(size_t)(m0 + row) * K + kk + q8]);
            cp16(&Bs[buf][row * 32 + q8], &Wt[(size_t)(n0 + row) * K + kk + q8]);
        }
    };

    stage(0, 0);
    int nk = K >> 5;
    for (int t = 0; t < nk; ++t) {
        async_fence();
        __syncthreads();
        int cur = t & 1;
        if (t + 1 < nk) stage(cur ^ 1, (t + 1) * 32);

        Frag a[2], b[4];
#pragma unroll
        for (int r = 0; r < 2; ++r)
#pragma unroll
            for (int v = 0; v < 8; ++v) {
                int kb = ((v & 4) ? 16 : 0) + hi * 8 + (v & 3) * 2;
                a[r].u[v] = *(const unsigned*)&As[cur][(wm * 32 + r * 16 + m16) * 32 + kb];
            }
#pragma unroll
        for (int c = 0; c < 4; ++c)
#pragma unroll
            for (int v = 0; v < 8; ++v) {
                int kb = ((v & 4) ? 16 : 0) + hi * 8 + (v & 3) * 2;
                b[c].u[v] = *(const unsigned*)&Bs[cur][(wn * 64 + c * 16 + m16) * 32 + kb];
            }
#pragma unroll
        for (int r = 0; r < 2; ++r)
#pragma unroll
            for (int c = 0; c < 4; ++c)
                acc[r][c] = __builtin_amdgcn_wmma_f32_16x16x32_bf16(
                    false, a[r].v, false, b[c].v, (short)0, acc[r][c], false, false);
    }
#pragma unroll
    for (int r = 0; r < 2; ++r)
#pragma unroll
        for (int c = 0; c < 4; ++c)
#pragma unroll
            for (int v = 0; v < 8; ++v) {
                int row = m0 + wm * 32 + r * 16 + hi * 8 + v;
                int col = n0 + wn * 64 + c * 16 + m16;
                float val = acc[r][c][v] + bias[col];
                if (RELU) val = fmaxf(val, 0.f);
                if (HAS_RES) val += res[(size_t)row * N + col];
                if (OUT_BF16) ((u16*)out)[(size_t)row * N + col] = f2bf(val);
                else          ((float*)out)[(size_t)row * N + col] = val;
            }
}

// ---------------------------------------------------------------- fused attention (per b,h, 32 q-rows)
// x1 = x + softmax(q k^T/8) * group_prob @ v ; V supplied pre-transposed vt[b][c][s]
__global__ void attn_kernel(const u16* __restrict__ q2, const u16* __restrict__ k2,
                            const u16* __restrict__ vt, const float* __restrict__ gp,
                            const float* __restrict__ xin, float* __restrict__ x1) {
    extern __shared__ char smem[];
    float* sc   = (float*)smem;                // [32][512]  65536 B
    float* red  = (float*)(smem + 65536);      // [32][8]     1024 B
    u16*   attb = (u16*)(smem + 66560);        // [32][512]  32768 B
    u16*   stg  = (u16*)(smem + 99328);        // [64][64]    8192 B
    u16*   qs   = (u16*)(smem + 107520);       // [32][64]    4096 B

    int b = blockIdx.z, h = blockIdx.y, qt = blockIdx.x;
    int q0 = qt * 32;
    int tid = threadIdx.x, wave = tid >> 5, lane = tid & 31;
    int m16 = lane & 15, hi = lane >> 4;
    int r = wave >> 2, c = wave & 3;

    // q tile [32][64]
    {
        int row = tid >> 3, off = (tid & 7) * 8;
        cp16(&qs[row * 64 + off],
             &q2[(size_t)(b * SEQ + q0 + row) * DMODEL + h * DKH + off]);
    }
    // ---- pass 1: scores ----
    for (int ck = 0; ck < 8; ++ck) {
        __syncthreads();
        {
            int row = tid >> 2, off = (tid & 3) * 16;
            const u16* src = &k2[(size_t)(b * SEQ + ck * 64 + row) * DMODEL + h * DKH + off];
            cp16(&stg[row * 64 + off], src);
            cp16(&stg[row * 64 + off + 8], src + 8);
        }
        async_fence();
        __syncthreads();
        v8f acc = (v8f){0,0,0,0,0,0,0,0};
#pragma unroll
        for (int ks = 0; ks < 2; ++ks) {
            Frag A, B;
#pragma unroll
            for (int v = 0; v < 8; ++v) {
                int kb = ((v & 4) ? 16 : 0) + hi * 8 + (v & 3) * 2 + ks * 32;
                A.u[v] = *(const unsigned*)&qs[(r * 16 + m16) * 64 + kb];
                B.u[v] = *(const unsigned*)&stg[(c * 16 + m16) * 64 + kb];
            }
            acc = __builtin_amdgcn_wmma_f32_16x16x32_bf16(
                false, A.v, false, B.v, (short)0, acc, false, false);
        }
#pragma unroll
        for (int v = 0; v < 8; ++v)
            sc[(r * 16 + hi * 8 + v) * 512 + ck * 64 + c * 16 + m16] = acc[v] * 0.125f;
    }
    __syncthreads();
    // ---- softmax * group_prob ----
    {
        int row = tid >> 3, sub = tid & 7;
        float mx = -1e30f;
        for (int i = 0; i < 64; ++i) mx = fmaxf(mx, sc[row * 512 + sub + i * 8]);
        red[row * 8 + sub] = mx; __syncthreads();
        float rmx = red[row * 8];
#pragma unroll
        for (int i = 1; i < 8; ++i) rmx = fmaxf(rmx, red[row * 8 + i]);
        __syncthreads();
        float sm = 0.f;
        for (int i = 0; i < 64; ++i) {
            int j = sub + i * 8;
            float p = __expf(sc[row * 512 + j] - rmx);
            sc[row * 512 + j] = p; sm += p;
        }
        red[row * 8 + sub] = sm; __syncthreads();
        float rsum = 0.f;
#pragma unroll
        for (int i = 0; i < 8; ++i) rsum += red[row * 8 + i];
        float inv = 1.f / rsum;
        size_t gbase = ((size_t)((b * NH + h) * SEQ + q0 + row)) * SEQ;
        for (int i = 0; i < 64; ++i) {
            int j = sub + i * 8;
            attb[row * 512 + j] = f2bf(sc[row * 512 + j] * inv * gp[gbase + j]);
        }
    }
    __syncthreads();
    // ---- pass 2: att @ v ----
    v8f acc = (v8f){0,0,0,0,0,0,0,0};
    for (int ck = 0; ck < 8; ++ck) {
        {
            int row = tid >> 2, off = (tid & 3) * 16;   // row = dk 0..63
            const u16* src = &vt[(size_t)(b * DMODEL + h * DKH + row) * SEQ + ck * 64 + off];
            cp16(&stg[row * 64 + off], src);
            cp16(&stg[row * 64 + off + 8], src + 8);
        }
        async_fence();
        __syncthreads();
#pragma unroll
        for (int ks = 0; ks < 2; ++ks) {
            Frag A, B;
#pragma unroll
            for (int v = 0; v < 8; ++v) {
                int kb = ((v & 4) ? 16 : 0) + hi * 8 + (v & 3) * 2 + ks * 32;
                A.u[v] = *(const unsigned*)&attb[(r * 16 + m16) * 512 + ck * 64 + kb];
                B.u[v] = *(const unsigned*)&stg[(c * 16 + m16) * 64 + kb];
            }
            acc = __builtin_amdgcn_wmma_f32_16x16x32_bf16(
                false, A.v, false, B.v, (short)0, acc, false, false);
        }
        __syncthreads();
    }
#pragma unroll
    for (int v = 0; v < 8; ++v) {
        int rq = q0 + r * 16 + hi * 8 + v;
        int col = h * DKH + c * 16 + m16;
        size_t idx = (size_t)(b * SEQ + rq) * DMODEL + col;
        x1[idx] = xin[idx] + acc[v];
    }
}

// ---------------------------------------------------------------- launcher
extern "C" void kernel_launch(void* const* d_in, const int* in_sizes, int n_in,
                              void* d_out, int out_size, void* d_ws, size_t ws_size,
                              hipStream_t stream) {
    const float* x     = (const float*)d_in[0];
    const float* prior = (const float*)d_in[2];
    const float* gn_g  = (const float*)d_in[3];
    const float* gn_b  = (const float*)d_in[4];
    const float* wq    = (const float*)d_in[5];
    const float* bq    = (const float*)d_in[6];
    const float* wk    = (const float*)d_in[7];
    const float* bk    = (const float*)d_in[8];
    const float* fq_w  = (const float*)d_in[9];
    const float* fq_b  = (const float*)d_in[10];
    const float* fk_w  = (const float*)d_in[11];
    const float* fk_b  = (const float*)d_in[12];
    const float* fv_w  = (const float*)d_in[13];
    const float* fv_b  = (const float*)d_in[14];
    const float* ln1_g = (const float*)d_in[15];
    const float* ln1_b = (const float*)d_in[16];
    const float* ff_w1 = (const float*)d_in[17];
    const float* ff_b1 = (const float*)d_in[18];
    const float* ff_w2 = (const float*)d_in[19];
    const float* ff_b2 = (const float*)d_in[20];
    const float* ln2_g = (const float*)d_in[21];
    const float* ln2_b = (const float*)d_in[22];

    float* out_x  = (float*)d_out;
    float* out_gp = out_x + (size_t)MROWS * DMODEL;
    float* out_na = out_gp + (size_t)BSZ * NH * SEQ * SEQ;

    char* wp = (char*)d_ws;
    auto carve = [&](size_t bytes) { void* p = wp; wp += (bytes + 255) & ~(size_t)255; return p; };
    u16*   ctxb = (u16*)carve((size_t)MROWS * DMODEL * 2);
    u16*   xnb  = (u16*)carve((size_t)MROWS * DMODEL * 2);
    u16*   fqwt = (u16*)carve((size_t)DMODEL * DMODEL * 2);
    u16*   fkwt = (u16*)carve((size_t)DMODEL * DMODEL * 2);
    u16*   fvwt = (u16*)carve((size_t)DMODEL * DMODEL * 2);
    u16*   w1t  = (u16*)carve((size_t)DMODEL * FFD * 2);
    u16*   w2t  = (u16*)carve((size_t)FFD * DMODEL * 2);
    u16*   q2b  = (u16*)carve((size_t)MROWS * DMODEL * 2);
    u16*   k2b  = (u16*)carve((size_t)MROWS * DMODEL * 2);
    u16*   v2b  = (u16*)carve((size_t)MROWS * DMODEL * 2);
    u16*   vtb  = (u16*)carve((size_t)MROWS * DMODEL * 2);
    float* qv   = (float*)carve((size_t)BSZ * NH * SEQ * DKH * 4);
    float* kv   = (float*)carve((size_t)BSZ * NH * SEQ * DKH * 4);
    float* naB  = (float*)carve((size_t)BSZ * NH * SEQ * 4);
    float* Pg   = (float*)carve((size_t)BSZ * NH * SEQ * 4);
    float* x1   = (float*)carve((size_t)MROWS * DMODEL * 4);
    u16*   xn2b = (u16*)carve((size_t)MROWS * DMODEL * 2);
    u16*   hb   = (u16*)carve((size_t)MROWS * FFD * 2);

    // 1) weight transpose-conversions: [K][N] f32 -> [N][K] bf16
    f2bf_t_kernel<<<dim3(DMODEL / 32, DMODEL / 32), 256, 0, stream>>>(fq_w, fqwt, DMODEL, DMODEL);
    f2bf_t_kernel<<<dim3(DMODEL / 32, DMODEL / 32), 256, 0, stream>>>(fk_w, fkwt, DMODEL, DMODEL);
    f2bf_t_kernel<<<dim3(DMODEL / 32, DMODEL / 32), 256, 0, stream>>>(fv_w, fvwt, DMODEL, DMODEL);
    f2bf_t_kernel<<<dim3(DMODEL / 32, FFD / 32), 256, 0, stream>>>(ff_w1, w1t, DMODEL, FFD);
    f2bf_t_kernel<<<dim3(FFD / 32, DMODEL / 32), 256, 0, stream>>>(ff_w2, w2t, FFD, DMODEL);

    // 2) dual LN: ctx (gn) and xn (ln1)
    ln_dual_kernel<<<MROWS, 256, 0, stream>>>(x, gn_g, gn_b, ln1_g, ln1_b, ctxb, xnb);

    // 3) per-head q/k band projections
    qkproj_kernel<<<(BSZ * NH * SEQ * DKH) / 256, 256, 0, stream>>>(ctxb, wq, bq, wk, bk, qv, kv);

    // 4) band softmax + prefix scan
    band_kernel<<<BSZ * NH, 512, 0, stream>>>(qv, kv, prior, naB, Pg);

    // 5) group_prob + na fill
    fill_kernel<<<BSZ * NH * SEQ, 512, 0, stream>>>(naB, Pg, prior, out_gp, out_na);

    // 6) q2/k2/v2 projections (WMMA)
    dim3 gP(DMODEL / 128, MROWS / 128);
    gemm_bf16<0, 0, 1><<<gP, 256, 0, stream>>>(xnb, fqwt, fq_b, nullptr, q2b, MROWS, DMODEL, DMODEL);
    gemm_bf16<0, 0, 1><<<gP, 256, 0, stream>>>(xnb, fkwt, fk_b, nullptr, k2b, MROWS, DMODEL, DMODEL);
    gemm_bf16<0, 0, 1><<<gP, 256, 0, stream>>>(xnb, fvwt, fv_b, nullptr, v2b, MROWS, DMODEL, DMODEL);

    // 6b) pre-transpose V for async row-contiguous staging
    vtrans_kernel<<<dim3(SEQ / 32, DMODEL / 32, BSZ), 256, 0, stream>>>(v2b, vtb);

    // 7) attention: x1 = x + sa
    attn_kernel<<<dim3(SEQ / 32, NH, BSZ), 256, 111616, stream>>>(q2b, k2b, vtb, out_gp, x, x1);

    // 8) LN2
    ln_dual_kernel<<<MROWS, 256, 0, stream>>>(x1, ln2_g, ln2_b, ln2_g, ln2_b, xn2b, nullptr);

    // 9) FFN1 (relu)
    gemm_bf16<1, 0, 1><<<dim3(FFD / 128, MROWS / 128), 256, 0, stream>>>(
        xn2b, w1t, ff_b1, nullptr, hb, MROWS, FFD, DMODEL);

    // 10) FFN2 + residual -> out x (fp32)
    gemm_bf16<0, 1, 0><<<dim3(DMODEL / 128, MROWS / 128), 256, 0, stream>>>(
        hb, w2t, ff_b2, x1, out_x, MROWS, DMODEL, FFD);
}
